// BiMamba2Block_1640677507601
// MI455X (gfx1250) — compile-verified, matrix operand order checked by reading, not run on hardware
//
#include <hip/hip_runtime.h>
#include <hip/hip_bf16.h>
#include <cstdint>

#define DM      128
#define DSTATE  64
#define DINNER  256
#define NH      8
#define HD      32
#define CONVDIM 384
#define NPROJ   648
#define NPROJP  656     // padded to multiple of 16
#define KFRONT  384     // 3 shifted taps * 128
#define BB      16
#define LL      8192
#define NTOK    (BB*LL)
#define CHUNK   64
#define NCH     (LL/CHUNK)   // 128
#define EPS_F   1e-5f

typedef __attribute__((ext_vector_type(16))) _Float16 v16h;
typedef __attribute__((ext_vector_type(8)))  float    v8f;

// ---- WMMA fragment helpers (CDNA5 16x16x32 f16 layouts, CDNA5 ISA §7.12.2) ----
__device__ __forceinline__ int kmap(int h, int grp) {
  return (h & 7) + ((h >> 3) << 4) + (grp << 3);
}
__device__ __forceinline__ v16h frag_nk(const _Float16* __restrict__ src, int row, int ld,
                                        int k0, int grp) {
  v16h a;
#pragma unroll
  for (int h = 0; h < 16; ++h) a[h] = src[row * ld + k0 + kmap(h, grp)];
  return a;
}
__device__ __forceinline__ v16h frag_kn(const _Float16* __restrict__ src, int col, int ld,
                                        int k0, int grp) {
  v16h a;
#pragma unroll
  for (int h = 0; h < 16; ++h) a[h] = src[(k0 + kmap(h, grp)) * ld + col];
  return a;
}
__device__ __forceinline__ v8f wmma_f16(v16h a, v16h b, v8f c) {
  return __builtin_amdgcn_wmma_f32_16x16x32_f16(false, a, false, b, (short)0, c, false, false);
}
__device__ __forceinline__ float siluf(float v) { return v / (1.f + __expf(-v)); }

// ---- CDNA5 async global->LDS DMA (ASYNCcnt path, CDNA5 ISA) ----
__device__ __forceinline__ void async_load_b128(uint32_t lds_off, const void* gaddr) {
  asm volatile("global_load_async_to_lds_b128 %0, %1, off"
               :: "v"(lds_off), "v"((uint64_t)(uintptr_t)gaddr)
               : "memory");
}
__device__ __forceinline__ void wait_async() {
  asm volatile("s_wait_asynccnt 0" ::: "memory");
}
__device__ __forceinline__ uint32_t lds_off_of(const void* p) {
  return (uint32_t)(uintptr_t)p;   // flat LDS aperture: low 32 bits are the LDS address
}

// ---------------- prep kernels ----------------
__global__ void f32_to_f16(const float* __restrict__ in, _Float16* __restrict__ out, int n) {
  int i = blockIdx.x * blockDim.x + threadIdx.x;
  if (i < n) out[i] = (_Float16)in[i];
}

// R = in_w @ flin_w   [648 x 128]
__global__ void compute_R(const float* __restrict__ in_w, const float* __restrict__ lin_w,
                          float* __restrict__ R) {
  int i = blockIdx.x * blockDim.x + threadIdx.x;
  if (i >= NPROJ * DM) return;
  int n = i / DM, c = i % DM;
  float s = 0.f;
  for (int j = 0; j < DM; ++j) s += in_w[n * DM + j] * lin_w[j * DM + c];
  R[i] = s;
}

// Fused front weight Wq[656][384] (f16): tap s in {0,1,2}; center tap includes identity (x + conv)
__global__ void compute_Wq(const float* __restrict__ R, const float* __restrict__ conv3_w,
                           _Float16* __restrict__ Wq) {
  int i = blockIdx.x * blockDim.x + threadIdx.x;
  if (i >= NPROJP * KFRONT) return;
  int n = i / KFRONT, k = i % KFRONT;
  int s = k / DM, c = k % DM;
  float v = 0.f;
  if (n < NPROJ) {
    for (int o = 0; o < DM; ++o) v += R[n * DM + o] * conv3_w[o * (DM * 3) + c * 3 + s];
    if (s == 1) v += R[n * DM + c];
  }
  Wq[i] = (_Float16)v;
}

__global__ void compute_qbias(const float* __restrict__ R, const float* __restrict__ in_w,
                              const float* __restrict__ conv_b, const float* __restrict__ lin_b,
                              float* __restrict__ qbias) {
  int n = blockIdx.x * blockDim.x + threadIdx.x;
  if (n >= NPROJP) return;
  float v = 0.f;
  if (n < NPROJ)
    for (int j = 0; j < DM; ++j) v += R[n*DM+j]*conv_b[j] + in_w[n*DM+j]*lin_b[j];
  qbias[n] = v;
}

// ---------------- fused front GEMM: zxbcdt = fused(conv+lin+in_proj)(x), optional reversal ----------------
__global__ void __launch_bounds__(128)
front_gemm(const _Float16* __restrict__ xh, const _Float16* __restrict__ Wq,
           const float* __restrict__ qbias, _Float16* __restrict__ zxbcdt, int rev)
{
  __shared__ _Float16 Xs[16 * KFRONT];   // 16 tokens x (3 shifted rows of 128)
  const int tile = blockIdx.x;
  const int b  = tile / (LL / 16);
  const int t0 = (tile % (LL / 16)) * 16;
  const int tid = threadIdx.x, lane = tid & 31, wave = tid >> 5;
  const int grp = lane >> 4, col = lane & 15;

  // async-stage 48 shifted 256B rows as 16B units; zero-fill boundary rows via LDS stores
  for (int u = tid; u < 16 * 3 * 16; u += 128) {
    int seg = u >> 4, part = u & 15;
    int m = seg / 3, s = seg % 3;
    int tl = t0 + m + (s - 1);
    char* dst = (char*)&Xs[m * KFRONT + s * DM] + part * 16;
    if (tl < 0 || tl >= LL) {
      uint32_t* d32 = (uint32_t*)dst;
      d32[0] = 0u; d32[1] = 0u; d32[2] = 0u; d32[3] = 0u;
    } else {
      int tg = rev ? (LL - 1 - tl) : tl;
      const char* src = (const char*)(xh + (size_t)(b * LL + tg) * DM) + part * 16;
      async_load_b128(lds_off_of(dst), src);
    }
  }
  wait_async();
  __syncthreads();

  for (int nt = wave; nt < NPROJP / 16; nt += 4) {
    const int n = nt * 16 + col;
    v8f acc = {};
#pragma unroll
    for (int ks = 0; ks < KFRONT; ks += 32) {
      v16h a = frag_nk(Xs, col, KFRONT, ks, grp);
      v16h w = frag_nk(Wq, n, KFRONT, ks, grp);
      acc = wmma_f16(a, w, acc);
    }
    float qb = qbias[n];
#pragma unroll
    for (int r = 0; r < 8; ++r) {
      int m = r + 8 * grp;
      zxbcdt[(size_t)(b * LL + t0 + m) * NPROJP + n] = (_Float16)(acc[r] + qb);
    }
  }
}

// ---------------- depthwise causal conv (K=4) + silu ----------------
__global__ void dconv_silu(const _Float16* __restrict__ zxbcdt, const float* __restrict__ cw,
                           const float* __restrict__ cb, _Float16* __restrict__ xBC)
{
  int i = blockIdx.x * blockDim.x + threadIdx.x;
  if (i >= NTOK * CONVDIM) return;
  int t = i / CONVDIM, c = i % CONVDIM;
  int tl = t % LL;
  float s = cb[c];
#pragma unroll
  for (int k = 0; k < 4; ++k) {
    int tt = tl - 3 + k;
    if (tt >= 0) s += cw[c*4+k] * (float)zxbcdt[(size_t)(t - 3 + k) * NPROJP + DINNER + c];
  }
  xBC[i] = (_Float16)siluf(s);
}

__global__ void dt_softplus(const _Float16* __restrict__ zxbcdt, const float* __restrict__ dt_bias,
                            float* __restrict__ dtb) {
  int i = blockIdx.x*blockDim.x + threadIdx.x;
  if (i >= NTOK*NH) return;
  int t = i / NH, h = i % NH;
  float v = (float)zxbcdt[(size_t)t*NPROJP + (NPROJ - NH) + h] + dt_bias[h];
  dtb[i] = (v > 20.f) ? v : log1pf(__expf(v));
}

// ---------------- chunked SSD scan: intra-chunk (WMMA) ----------------
__global__ void __launch_bounds__(128)
chunk_intra(const _Float16* __restrict__ xBC, const float* __restrict__ dtb,
            const float* __restrict__ A_log, float* __restrict__ ybuf,
            float* __restrict__ Sc, float* __restrict__ cdecay)
{
  __shared__ _Float16 Bs[CHUNK*DSTATE], Cs2[CHUNK*DSTATE], Xs[CHUNK*HD], Ms[CHUNK*CHUNK];
  __shared__ float cs[CHUNK], dts[CHUNK];
  const int blk = blockIdx.x;
  const int c = blk % NCH, h = (blk / NCH) % NH, b = blk / (NCH*NH);
  const int T0 = b*LL + c*CHUNK;
  const int tid = threadIdx.x, lane = tid&31, wave = tid>>5, grp = lane>>4, col = lane&15;

  // async-stage B, C (64x64 f16) and X (64x32 f16) tiles in 16B units
  for (int u = tid; u < (CHUNK*DSTATE)/8; u += 128) {
    int r = u >> 3, n8 = (u & 7) * 8;
    async_load_b128(lds_off_of(&Bs[r*DSTATE + n8]),
                    &xBC[(size_t)(T0+r)*CONVDIM + DINNER + n8]);
    async_load_b128(lds_off_of(&Cs2[r*DSTATE + n8]),
                    &xBC[(size_t)(T0+r)*CONVDIM + DINNER + DSTATE + n8]);
  }
  for (int u = tid; u < (CHUNK*HD)/8; u += 128) {
    int r = u >> 2, p8 = (u & 3) * 8;
    async_load_b128(lds_off_of(&Xs[r*HD + p8]),
                    &xBC[(size_t)(T0+r)*CONVDIM + h*HD + p8]);
  }
  if (tid == 0) {
    float A = -__expf(A_log[h]);
    float run = 0.f;
    for (int j = 0; j < CHUNK; ++j) {
      float d = dtb[(size_t)(T0+j)*NH + h];
      dts[j] = d; run += d * A; cs[j] = run;
    }
  }
  wait_async();
  __syncthreads();

  // G = C @ B^T (K = 64 states); masked by causal decay -> Ms (f16)
  for (int tidx = wave; tidx < 16; tidx += 4) {
    int mt = tidx >> 2, nt = tidx & 3;
    v8f acc = {};
#pragma unroll
    for (int ks = 0; ks < DSTATE; ks += 32) {
      v16h a   = frag_nk(Cs2, mt*16 + col, DSTATE, ks, grp);
      v16h bfr = frag_nk(Bs,  nt*16 + col, DSTATE, ks, grp);
      acc = wmma_f16(a, bfr, acc);
    }
#pragma unroll
    for (int r = 0; r < 8; ++r) {
      int m = mt*16 + r + 8*grp, n = nt*16 + col;
      float lmask = (m >= n) ? __expf(cs[m]-cs[n]) * dts[n] : 0.f;
      Ms[m*CHUNK + n] = (_Float16)(acc[r]*lmask);
    }
  }
  __syncthreads();

  // Y_intra = Ms @ X  [64 x 32]
  for (int tidx = wave; tidx < 8; tidx += 4) {
    int mt = tidx & 3, nt = tidx >> 2;
    v8f acc = {};
#pragma unroll
    for (int ks = 0; ks < CHUNK; ks += 32) {
      v16h a   = frag_nk(Ms, mt*16 + col, CHUNK, ks, grp);
      v16h bfr = frag_kn(Xs, nt*16 + col, HD, ks, grp);
      acc = wmma_f16(a, bfr, acc);
    }
#pragma unroll
    for (int r = 0; r < 8; ++r) {
      int m = mt*16 + r + 8*grp, p = nt*16 + col;
      ybuf[(size_t)(T0+m)*DINNER + h*HD + p] = acc[r];
    }
  }
  __syncthreads();

  // Bw^T into Ms: Ms[n][i] = B[i][n] * exp(cs_total - cs[i]) * dt[i]
  for (int i = tid; i < DSTATE*CHUNK; i += 128) {
    int n = i >> 6, ii = i & 63;
    float w = __expf(cs[CHUNK-1]-cs[ii]) * dts[ii];
    Ms[i] = (_Float16)((float)Bs[ii*DSTATE + n] * w);
  }
  __syncthreads();

  // chunk state contribution S_c = Bw^T @ X  [64 states x 32]
  float* ScOut = Sc + (size_t)blk * (DSTATE*HD);
  for (int tidx = wave; tidx < 8; tidx += 4) {
    int mt = tidx & 3, nt = tidx >> 2;
    v8f acc = {};
#pragma unroll
    for (int ks = 0; ks < CHUNK; ks += 32) {
      v16h a   = frag_nk(Ms, mt*16 + col, CHUNK, ks, grp);
      v16h bfr = frag_kn(Xs, nt*16 + col, HD, ks, grp);
      acc = wmma_f16(a, bfr, acc);
    }
#pragma unroll
    for (int r = 0; r < 8; ++r) {
      int n = mt*16 + r + 8*grp, p = nt*16 + col;
      ScOut[n*HD + p] = acc[r];
    }
  }
  if (tid == 0) cdecay[blk] = __expf(cs[CHUNK-1]);
}

// ---------------- sequential chunk-state carry (tiny: 128 steps over 64x32 per (b,h)) ----------------
__global__ void __launch_bounds__(256)
chunk_carry(float* __restrict__ Sc, const float* __restrict__ cdecay) {
  int bh = blockIdx.x;
  int tid = threadIdx.x;
  float run[8];
#pragma unroll
  for (int k = 0; k < 8; ++k) run[k] = 0.f;
  for (int c = 0; c < NCH; ++c) {
    size_t base = ((size_t)bh * NCH + c) * (DSTATE*HD);
    float d = cdecay[bh*NCH + c];
#pragma unroll
    for (int k = 0; k < 8; ++k) {
      size_t e = base + (size_t)tid*8 + k;
      float contrib = Sc[e];
      Sc[e] = run[k];                 // in-place: prefix state entering chunk c
      run[k] = run[k]*d + contrib;
    }
  }
}

// ---------------- inter-chunk: Y += exp(cs)*C@S_prefix + D*x (WMMA) ----------------
__global__ void __launch_bounds__(128)
chunk_inter(const _Float16* __restrict__ xBC, const float* __restrict__ dtb,
            const float* __restrict__ A_log, const float* __restrict__ Sp,
            const float* __restrict__ Dvec, float* __restrict__ ybuf)
{
  __shared__ _Float16 Cs2[CHUNK*DSTATE], Sps[DSTATE*HD], Xs[CHUNK*HD];
  __shared__ float cs[CHUNK];
  const int blk = blockIdx.x;
  const int c = blk % NCH, h = (blk / NCH) % NH, b = blk / (NCH*NH);
  const int T0 = b*LL + c*CHUNK;
  const int tid = threadIdx.x, lane = tid&31, wave = tid>>5, grp = lane>>4, col = lane&15;

  for (int u = tid; u < (CHUNK*DSTATE)/8; u += 128) {
    int r = u >> 3, n8 = (u & 7) * 8;
    async_load_b128(lds_off_of(&Cs2[r*DSTATE + n8]),
                    &xBC[(size_t)(T0+r)*CONVDIM + DINNER + DSTATE + n8]);
  }
  for (int u = tid; u < (CHUNK*HD)/8; u += 128) {
    int r = u >> 2, p8 = (u & 3) * 8;
    async_load_b128(lds_off_of(&Xs[r*HD + p8]),
                    &xBC[(size_t)(T0+r)*CONVDIM + h*HD + p8]);
  }
  const float* spg = Sp + (size_t)blk * (DSTATE*HD);
  for (int i = tid; i < DSTATE*HD; i += 128) Sps[i] = (_Float16)spg[i];
  if (tid == 0) {
    float A = -__expf(A_log[h]);
    float run = 0.f;
    for (int j = 0; j < CHUNK; ++j) { run += dtb[(size_t)(T0+j)*NH + h] * A; cs[j] = run; }
  }
  wait_async();
  __syncthreads();

  float Dh = Dvec[h];
  for (int tidx = wave; tidx < 8; tidx += 4) {
    int mt = tidx & 3, nt = tidx >> 2;
    v8f acc = {};
#pragma unroll
    for (int ks = 0; ks < DSTATE; ks += 32) {
      v16h a   = frag_nk(Cs2, mt*16 + col, DSTATE, ks, grp);
      v16h bfr = frag_kn(Sps, nt*16 + col, HD, ks, grp);
      acc = wmma_f16(a, bfr, acc);
    }
#pragma unroll
    for (int r = 0; r < 8; ++r) {
      int m = mt*16 + r + 8*grp, p = nt*16 + col;
      size_t yi = (size_t)(T0+m)*DINNER + h*HD + p;
      ybuf[yi] += acc[r]*__expf(cs[m]) + Dh * (float)Xs[m*HD + p];
    }
  }
}

// ---------------- gated RMSNorm over 256 (one wave per token) ----------------
__global__ void __launch_bounds__(256)
gate_rms(const float* __restrict__ ybuf, const _Float16* __restrict__ zxbcdt,
         const float* __restrict__ norm_w, _Float16* __restrict__ yn)
{
  int warp = (blockIdx.x * blockDim.x + threadIdx.x) >> 5;
  int lane = threadIdx.x & 31;
  if (warp >= NTOK) return;
  size_t t = (size_t)warp;
  float v[8]; float ss = 0.f;
#pragma unroll
  for (int k = 0; k < 8; ++k) {
    int cdx = lane + 32*k;
    float z = (float)zxbcdt[t*NPROJP + cdx];
    float val = ybuf[t*DINNER + cdx] * siluf(z);
    v[k] = val; ss += val*val;
  }
  for (int off = 16; off > 0; off >>= 1) ss += __shfl_down(ss, off);
  ss = __shfl(ss, 0);
  float inv = rsqrtf(ss/(float)DINNER + EPS_F);
#pragma unroll
  for (int k = 0; k < 8; ++k) {
    int cdx = lane + 32*k;
    yn[t*DINNER + cdx] = (_Float16)(v[k]*inv*norm_w[cdx]);
  }
}

// ---------------- generic token-GEMM (16-token tiles, WMMA) ----------------
#define GF_SILU   1
#define GF_OUTF16 2
#define GF_ACC    4
#define GF_REV    8
__global__ void __launch_bounds__(128)
wmma_gemm(const _Float16* __restrict__ X, const _Float16* __restrict__ W,
          const float* __restrict__ bias, float* __restrict__ outf,
          _Float16* __restrict__ outh, int K, int N, int flags)
{
  __shared__ _Float16 Xs[16*KFRONT];
  const int tile = blockIdx.x;
  const int b  = tile / (LL/16);
  const int t0 = (tile % (LL/16))*16;
  const int tid = threadIdx.x, lane=tid&31, wave=tid>>5, grp=lane>>4, col=lane&15;

  const char* src = (const char*)(X + (size_t)(b*LL+t0)*K);
  char* dstl = (char*)Xs;
  const int nbytes = 16*K*2;
  for (int o = tid*16; o < nbytes; o += 128*16)
    async_load_b128(lds_off_of(dstl + o), src + o);
  wait_async();
  __syncthreads();

  for (int nt = wave; nt < N/16; nt += 4) {
    int n = nt*16 + col;
    v8f acc = {};
    for (int ks = 0; ks < K; ks += 32) {
      v16h a = frag_nk(Xs, col, K, ks, grp);
      v16h w = frag_nk(W, n, K, ks, grp);
      acc = wmma_f16(a, w, acc);
    }
    float bv = bias ? bias[n] : 0.f;
#pragma unroll
    for (int r = 0; r < 8; ++r) {
      int m = r + 8*grp;
      int tl = t0 + m;
      int to = (flags & GF_REV) ? (LL-1-tl) : tl;
      size_t idx = (size_t)(b*LL+to)*N + n;
      float v = acc[r] + bv;
      if (flags & GF_SILU) v = siluf(v);
      if (flags & GF_OUTF16)      outh[idx] = (_Float16)v;
      else if (flags & GF_ACC)    outf[idx] += v;
      else                        outf[idx] = v;
    }
  }
}

__global__ void gmul(const float* __restrict__ acc, const _Float16* __restrict__ gate,
                     _Float16* __restrict__ t1) {
  int i = blockIdx.x*blockDim.x + threadIdx.x;
  if (i < NTOK*DM) t1[i] = (_Float16)(acc[i] * (float)gate[i]);
}

__global__ void __launch_bounds__(256)
final_rms(const float* __restrict__ o, const float* __restrict__ x,
          const float* __restrict__ norm_w, float* __restrict__ out)
{
  int warp = (blockIdx.x*blockDim.x + threadIdx.x) >> 5;
  int lane = threadIdx.x & 31;
  if (warp >= NTOK) return;
  size_t t = (size_t)warp;
  float v[4]; float ss = 0.f;
#pragma unroll
  for (int k = 0; k < 4; ++k) { float val = o[t*DM + lane + 32*k]; v[k]=val; ss += val*val; }
  for (int off=16; off>0; off>>=1) ss += __shfl_down(ss, off);
  ss = __shfl(ss, 0);
  float inv = rsqrtf(ss/(float)DM + EPS_F);
#pragma unroll
  for (int k = 0; k < 4; ++k) {
    int cdx = lane + 32*k;
    out[t*DM+cdx] = x[t*DM+cdx] + v[k]*inv*norm_w[cdx];
  }
}

// ---------------- host orchestration ----------------
extern "C" void kernel_launch(void* const* d_in, const int* in_sizes, int n_in,
                              void* d_out, int out_size, void* d_ws, size_t ws_size,
                              hipStream_t stream)
{
  const float* x       = (const float*)d_in[0];
  const float* gate_w  = (const float*)d_in[1];
  const float* gate_b  = (const float*)d_in[2];
  const float* fconv_w = (const float*)d_in[3];
  const float* fconv_b = (const float*)d_in[4];
  const float* flin_w  = (const float*)d_in[5];
  const float* flin_b  = (const float*)d_in[6];
  const float* f_in_w  = (const float*)d_in[7];
  const float* f_cw    = (const float*)d_in[8];
  const float* f_cb    = (const float*)d_in[9];
  const float* f_dtb   = (const float*)d_in[10];
  const float* f_Alog  = (const float*)d_in[11];
  const float* f_D     = (const float*)d_in[12];
  const float* f_nw    = (const float*)d_in[13];
  const float* f_ow    = (const float*)d_in[14];
  const float* bconv_w = (const float*)d_in[15];
  const float* bconv_b = (const float*)d_in[16];
  const float* blin_w  = (const float*)d_in[17];
  const float* blin_b  = (const float*)d_in[18];
  const float* b_in_w  = (const float*)d_in[19];
  const float* b_cw    = (const float*)d_in[20];
  const float* b_cb    = (const float*)d_in[21];
  const float* b_dtb   = (const float*)d_in[22];
  const float* b_Alog  = (const float*)d_in[23];
  const float* b_D     = (const float*)d_in[24];
  const float* b_nw    = (const float*)d_in[25];
  const float* b_ow    = (const float*)d_in[26];
  const float* out_w   = (const float*)d_in[27];
  const float* out_b   = (const float*)d_in[28];
  const float* norm_w  = (const float*)d_in[29];

  char* p = (char*)d_ws;
  auto carve = [&](size_t bytes) { char* r = p; p += (bytes + 255) & ~(size_t)255; return r; };
  _Float16* xh16   = (_Float16*)carve((size_t)NTOK*DM*2);
  _Float16* gate16 = (_Float16*)carve((size_t)NTOK*DM*2);
  float*    accb   = (float*)   carve((size_t)NTOK*DM*4);
  _Float16* zx     = (_Float16*)carve((size_t)NTOK*NPROJP*2);
  float*    dtbuf  = (float*)   carve((size_t)NTOK*NH*4);
  _Float16* xBC    = (_Float16*)carve((size_t)NTOK*CONVDIM*2);
  float*    ybuf   = (float*)   carve((size_t)NTOK*DINNER*4);
  float*    Scbuf  = (float*)   carve((size_t)BB*NH*NCH*DSTATE*HD*4);
  float*    cdec   = (float*)   carve((size_t)BB*NH*NCH*4);
  float*    Rbuf   = (float*)   carve((size_t)NPROJ*DM*4);
  _Float16* Wq     = (_Float16*)carve((size_t)NPROJP*KFRONT*2);
  float*    qbias  = (float*)   carve((size_t)NPROJP*4);
  _Float16* Wg16   = (_Float16*)carve((size_t)DM*DM*2);
  _Float16* Wo16   = (_Float16*)carve((size_t)DM*DINNER*2);
  _Float16* Wf16   = (_Float16*)carve((size_t)DM*DM*2);
  _Float16* t1     = (_Float16*)carve((size_t)NTOK*DM*2);
  float*    obuf   = (float*)   carve((size_t)NTOK*DM*4);
  _Float16* yn16   = (_Float16*)carve((size_t)NTOK*DINNER*2);
  (void)ws_size; (void)in_sizes; (void)n_in; (void)out_size;

  dim3 b256(256);
  auto cvt = [&](const float* src, _Float16* dst, int n) {
    f32_to_f16<<<(n+255)/256, b256, 0, stream>>>(src, dst, n);
  };
  cvt(x, xh16, NTOK*DM);
  cvt(gate_w, Wg16, DM*DM);
  cvt(out_w, Wf16, DM*DM);

  // gate = silu(x @ gate_w.T + gate_b)
  wmma_gemm<<<NTOK/16, 128, 0, stream>>>(xh16, Wg16, gate_b, nullptr, gate16,
                                         DM, DM, GF_SILU|GF_OUTF16);

  auto direction = [&](const float* in_w, const float* cv3w, const float* cv3b,
                       const float* linw, const float* linb,
                       const float* cw, const float* cb, const float* dtbias,
                       const float* Alog, const float* Dv, const float* nw,
                       const float* ow, int rev) {
    compute_R    <<<(NPROJ*DM+255)/256, b256, 0, stream>>>(in_w, linw, Rbuf);
    compute_Wq   <<<(NPROJP*KFRONT+255)/256, b256, 0, stream>>>(Rbuf, cv3w, Wq);
    compute_qbias<<<(NPROJP+255)/256, b256, 0, stream>>>(Rbuf, in_w, cv3b, linb, qbias);
    front_gemm   <<<NTOK/16, 128, 0, stream>>>(xh16, Wq, qbias, zx, rev);
    dconv_silu   <<<(NTOK*CONVDIM+255)/256, b256, 0, stream>>>(zx, cw, cb, xBC);
    dt_softplus  <<<(NTOK*NH+255)/256, b256, 0, stream>>>(zx, dtbias, dtbuf);
    chunk_intra  <<<BB*NH*NCH, 128, 0, stream>>>(xBC, dtbuf, Alog, ybuf, Scbuf, cdec);
    chunk_carry  <<<BB*NH, 256, 0, stream>>>(Scbuf, cdec);
    chunk_inter  <<<BB*NH*NCH, 128, 0, stream>>>(xBC, dtbuf, Alog, Scbuf, Dv, ybuf);
    gate_rms     <<<NTOK/8, b256, 0, stream>>>(ybuf, zx, nw, yn16);
    cvt(ow, Wo16, DM*DINNER);
    wmma_gemm<<<NTOK/16, 128, 0, stream>>>(yn16, Wo16, nullptr, accb, nullptr,
                                           DINNER, DM, rev ? (GF_ACC|GF_REV) : 0);
  };
  direction(f_in_w, fconv_w, fconv_b, flin_w, flin_b, f_cw, f_cb, f_dtb, f_Alog, f_D, f_nw, f_ow, 0);
  direction(b_in_w, bconv_w, bconv_b, blin_w, blin_b, b_cw, b_cb, b_dtb, b_Alog, b_D, b_nw, b_ow, 1);

  gmul<<<(NTOK*DM+255)/256, b256, 0, stream>>>(accb, gate16, t1);
  wmma_gemm<<<NTOK/16, 128, 0, stream>>>(t1, Wf16, out_b, obuf, nullptr, DM, DM, 0);
  final_rms<<<NTOK/8, b256, 0, stream>>>(obuf, x, norm_w, (float*)d_out);
}
